// DeformableBlock_37907381354832
// MI455X (gfx1250) — matrix operand order repaired and checked by reading, not compiled, not run on hardware
//
#include <hip/hip_runtime.h>
#include <hip/hip_bf16.h>

typedef __attribute__((ext_vector_type(16))) __bf16 v16bf;
typedef __attribute__((ext_vector_type(8)))  float  v8f;

#define BB   8
#define CIN  64
#define COUT 64
#define HH   96
#define WW   96
#define HW   (HH*WW)        // 9216
#define KK   9
#define KDIM (CIN*KK)       // 576
#define NPIX (BB*HW)        // 73728
#define KSTEPS (KDIM/32)    // 18

__device__ __forceinline__ unsigned short f32_to_bf16(float f) {
    unsigned int u = __float_as_uint(f);
    u += 0x7FFFu + ((u >> 16) & 1u);   // round-to-nearest-even
    return (unsigned short)(u >> 16);
}

// ---------------------------------------------------------------------------
// Kernel 1: offset conv, direct 3x3, weights staged in LDS.
// ---------------------------------------------------------------------------
__global__ __launch_bounds__(256) void offset_conv_kernel(
    const float* __restrict__ x, const float* __restrict__ wof,
    const float* __restrict__ bof, float* __restrict__ off)
{
    __shared__ float sw[18 * CIN * KK];   // 41472 B
    for (int i = threadIdx.x; i < 18 * CIN * KK; i += 256) sw[i] = wof[i];
    __syncthreads();

    const int p  = blockIdx.x * 256 + threadIdx.x;     // NPIX = 288*256 exact
    const int b  = p / HW;
    const int hw = p % HW;
    const int h  = hw / WW, w = hw % WW;

    float acc[18];
#pragma unroll
    for (int o = 0; o < 18; ++o) acc[o] = 0.f;

    for (int c = 0; c < CIN; ++c) {
        float xv[9];
#pragma unroll
        for (int t = 0; t < 9; ++t) {
            const int yy = h + t / 3 - 1, xx = w + t % 3 - 1;
            xv[t] = (yy >= 0 && yy < HH && xx >= 0 && xx < WW)
                        ? x[(((size_t)b * CIN + c) * HH + yy) * WW + xx] : 0.f;
        }
#pragma unroll
        for (int o = 0; o < 18; ++o) {
            const float* wl = &sw[(o * CIN + c) * KK];
#pragma unroll
            for (int t = 0; t < 9; ++t) acc[o] += xv[t] * wl[t];
        }
    }
#pragma unroll
    for (int o = 0; o < 18; ++o)
        off[((size_t)b * 18 + o) * HW + hw] = acc[o] + bof[o];
}

// ---------------------------------------------------------------------------
// Kernel 2: w_def (Cout, Cin*9) fp32 -> bf16 (row-major over K = c*9+tap)
// ---------------------------------------------------------------------------
__global__ void wdef_to_bf16_kernel(const float* __restrict__ wd,
                                    unsigned short* __restrict__ wbf)
{
    const int i = blockIdx.x * 256 + threadIdx.x;
    if (i < COUT * KDIM) wbf[i] = f32_to_bf16(wd[i]);
}

// ---------------------------------------------------------------------------
// Kernel 3: zero BN accumulators
// ---------------------------------------------------------------------------
__global__ void zero_kernel(float* __restrict__ p, int n)
{
    const int i = blockIdx.x * 256 + threadIdx.x;
    if (i < n) p[i] = 0.f;
}

// ---------------------------------------------------------------------------
// Kernel 4: fused deformable-sample + bf16 WMMA GEMM.
// WG = 128 threads = 4 waves; tile = 64 Cout x 32 row-aligned pixels.
// Each wave: 16 Cout rows x both 16-pixel halves (1 A frag -> 2 wmmas).
// Double-buffered LDS panel, one barrier per K-step; (c,tap) tracked
// incrementally (no integer division in the hot loop).
// ---------------------------------------------------------------------------
__global__ __launch_bounds__(128) void deform_gemm_kernel(
    const float* __restrict__ x,
    const float* __restrict__ off,
    const unsigned short* __restrict__ wbf,
    const float* __restrict__ b_def,
    float* __restrict__ ybuf)
{
    __shared__ float          sWt[288][4];        // masked bilinear weights
    __shared__ int            sIx[288][4];        // clamped flat hw indices
    __shared__ unsigned short sS[2][32][34];      // double-buffered [pix][k] panel

    const int tid     = threadIdx.x;
    const int pixBase = blockIdx.x * 32;
    const int b       = pixBase / HW;
    const int hw0     = pixBase % HW;
    const int h       = hw0 / WW;
    const int w0      = hw0 % WW;

    // Precompute bilinear corners for 9 taps x 32 pixels (torchvision OOB=0)
    for (int e = tid; e < 288; e += 128) {
        const int pix = e & 31;
        const int tap = e >> 5;
        const int ww  = w0 + pix;
        const float dy = off[(((size_t)b * 18 + 2 * tap)     * HH + h) * WW + ww];
        const float dx = off[(((size_t)b * 18 + 2 * tap + 1) * HH + h) * WW + ww];
        const float py = (float)(h  - 1 + tap / 3) + dy;
        const float px = (float)(ww - 1 + tap % 3) + dx;
        const float fy0 = floorf(py), fx0 = floorf(px);
        const float wy1 = py - fy0,  wx1 = px - fx0;
        const float wy0 = 1.f - wy1, wx0 = 1.f - wx1;
        const int y0 = (int)fy0, x0 = (int)fx0;
        const int y1 = y0 + 1,   x1 = x0 + 1;
        const bool vy0 = (y0 >= 0) & (y0 < HH), vy1 = (y1 >= 0) & (y1 < HH);
        const bool vx0 = (x0 >= 0) & (x0 < WW), vx1 = (x1 >= 0) & (x1 < WW);
        const int cy0 = y0 < 0 ? 0 : (y0 > HH - 1 ? HH - 1 : y0);
        const int cy1 = y1 < 0 ? 0 : (y1 > HH - 1 ? HH - 1 : y1);
        const int cx0 = x0 < 0 ? 0 : (x0 > WW - 1 ? WW - 1 : x0);
        const int cx1 = x1 < 0 ? 0 : (x1 > WW - 1 ? WW - 1 : x1);
        sWt[e][0] = (vy0 && vx0) ? wy0 * wx0 : 0.f;
        sWt[e][1] = (vy0 && vx1) ? wy0 * wx1 : 0.f;
        sWt[e][2] = (vy1 && vx0) ? wy1 * wx0 : 0.f;
        sWt[e][3] = (vy1 && vx1) ? wy1 * wx1 : 0.f;
        sIx[e][0] = cy0 * WW + cx0;
        sIx[e][1] = cy0 * WW + cx1;
        sIx[e][2] = cy1 * WW + cx0;
        sIx[e][3] = cy1 * WW + cx1;
    }

    const int lane     = tid & 31;
    const int wave     = tid >> 5;                 // 0..3
    const int coutBase = wave * 16;                // 4 Cout tiles of 16
    const int mrow     = lane & 15;
    const int hi       = lane >> 4;

    const int pixS = tid & 31;                     // sampling role: pixel
    const int kq   = tid >> 5;                     // sampling role: 8 k's each

    const float* xbb = x + (size_t)b * CIN * HW;

    // Incremental (c, tap) tracking for this thread's first k of each step:
    // seed for kidx = kq*8, then advance by 32 per step (32 = 3*9 + 5).
    int sc = (kq * 8) / 9;                         // one tiny div at entry
    int st = kq * 8 - sc * 9;

    // Sample a 32K x 32pix bf16 panel into buffer `buf`; (c0,tap0) = seed.
    auto sample_step = [&](int c0, int tap0, int buf) {
        int c = c0, tap = tap0;
#pragma unroll
        for (int j = 0; j < 8; ++j) {
            const int kl = kq * 8 + j;
            const int e  = tap * 32 + pixS;
            const float* xb = xbb + (size_t)c * HW;
            const float v = sWt[e][0] * xb[sIx[e][0]] + sWt[e][1] * xb[sIx[e][1]]
                          + sWt[e][2] * xb[sIx[e][2]] + sWt[e][3] * xb[sIx[e][3]];
            sS[buf][pixS][kl] = f32_to_bf16(v);
            ++tap;
            if (tap == 9) { tap = 0; ++c; }        // unrolled -> cmp+select
        }
    };
    auto advance32 = [&](int& c, int& tap) {
        c += 3; tap += 5;
        if (tap >= 9) { tap -= 9; ++c; }
    };

    __syncthreads();           // sWt/sIx visible
    sample_step(sc, st, 0);
    advance32(sc, st);
    __syncthreads();           // panel 0 ready

    v8f acc0 = {}, acc1 = {};

    for (int s = 0; s < KSTEPS; ++s) {
        const int buf   = s & 1;
        const int kBase = s * 32;

        // ---- fragments in documented 16-bit WMMA lane layouts ----
        union { v16bf v; unsigned int u[8]; } afrag, bfrag0, bfrag1;
        const unsigned short* wrow = wbf + (size_t)(coutBase + mrow) * KDIM + kBase;
#pragma unroll
        for (int v = 0; v < 8; ++v) {
            // A 16x32: VGPR v holds K = (v>=4?16:0) + hi*8 + (v&3)*2 .. +1
            const int ka = ((v >= 4) ? 16 : 0) + hi * 8 + (v & 3) * 2;
            afrag.u[v] = *(const unsigned int*)(wrow + ka);
            // B 32x16: lane = column, VGPR v holds K = (v>=4?16:0) + hi*8 + (v&3)*2 .. +1
            const int kb = ((v >= 4) ? 16 : 0) + hi * 8 + (v & 3) * 2;
            bfrag0.u[v] = *(const unsigned int*)&sS[buf][lane & 15][kb];
            bfrag1.u[v] = *(const unsigned int*)&sS[buf][16 + (lane & 15)][kb];
        }

        // keep next weight row warm at WGP level (global_prefetch_b8)
        if (s + 1 < KSTEPS)
            __builtin_prefetch((const void*)(wrow + 32), 0, 3);

        // overlap: gather next panel while this step's wmmas execute
        if (s + 1 < KSTEPS) {
            sample_step(sc, st, buf ^ 1);
            advance32(sc, st);
        }

        acc0 = __builtin_amdgcn_wmma_f32_16x16x32_bf16(
                   false, afrag.v, false, bfrag0.v, (short)0, acc0, false, false);
        acc1 = __builtin_amdgcn_wmma_f32_16x16x32_bf16(
                   false, afrag.v, false, bfrag1.v, (short)0, acc1, false, false);

        __syncthreads();       // panel buf^1 ready / buf free for rewrite
    }

    // ---- epilogue: D layout VGPR r -> M = r + hi*8, N = lane&15 ----
    const int n = lane & 15;
#pragma unroll
    for (int r = 0; r < 8; ++r) {
        const int ch = coutBase + r + hi * 8;
        const float bias = b_def[ch];
        float* yrow = ybuf + ((size_t)b * COUT + ch) * HW + hw0;
        yrow[n]      = acc0[r] + bias;
        yrow[n + 16] = acc1[r] + bias;
    }
}

// ---------------------------------------------------------------------------
// Kernel 5: per-channel sum/sumsq (one WG per (b,ch) slice of 9216 floats)
// ---------------------------------------------------------------------------
__global__ __launch_bounds__(256) void bn_stats_kernel(
    const float* __restrict__ y, float* __restrict__ sums, float* __restrict__ sumsq)
{
    __shared__ float s1[256], s2[256];
    const int slice = blockIdx.x;            // 0..511, layout (b,ch)
    const int ch    = slice & (COUT - 1);
    const float* yp = y + (size_t)slice * HW;
    float a = 0.f, q = 0.f;
    for (int i = threadIdx.x; i < HW; i += 256) {
        const float v = yp[i];
        a += v; q += v * v;
    }
    s1[threadIdx.x] = a; s2[threadIdx.x] = q;
    __syncthreads();
    for (int s = 128; s > 0; s >>= 1) {
        if (threadIdx.x < s) {
            s1[threadIdx.x] += s1[threadIdx.x + s];
            s2[threadIdx.x] += s2[threadIdx.x + s];
        }
        __syncthreads();
    }
    if (threadIdx.x == 0) {
        atomicAdd(&sums[ch],  s1[0]);
        atomicAdd(&sumsq[ch], s2[0]);
    }
}

// ---------------------------------------------------------------------------
// Kernel 6: in-place BN (biased var) + gamma/beta + ReLU
// ---------------------------------------------------------------------------
__global__ __launch_bounds__(256) void bn_relu_kernel(
    float* __restrict__ y, const float* __restrict__ sums,
    const float* __restrict__ sumsq, const float* __restrict__ gamma,
    const float* __restrict__ beta)
{
    const size_t idx = (size_t)blockIdx.x * 256 + threadIdx.x;
    const int ch = (int)((idx / HW) & (COUT - 1));
    const float N    = (float)(BB * HW);
    const float mean = sums[ch] / N;
    const float var  = sumsq[ch] / N - mean * mean;
    const float inv  = rsqrtf(var + 1e-5f) * gamma[ch];
    const float v    = (y[idx] - mean) * inv + beta[ch];
    y[idx] = v > 0.f ? v : 0.f;
}

// ---------------------------------------------------------------------------
extern "C" void kernel_launch(void* const* d_in, const int* in_sizes, int n_in,
                              void* d_out, int out_size, void* d_ws, size_t ws_size,
                              hipStream_t stream)
{
    const float* x     = (const float*)d_in[0];
    const float* w_off = (const float*)d_in[1];
    const float* b_off = (const float*)d_in[2];
    const float* w_def = (const float*)d_in[3];
    const float* b_def = (const float*)d_in[4];
    const float* gamma = (const float*)d_in[5];
    const float* beta  = (const float*)d_in[6];

    char* ws = (char*)d_ws;
    float*          off   = (float*)(ws + 0);                   // 8*18*9216 f32 = 5308416 B
    unsigned short* wbf   = (unsigned short*)(ws + 5308416);    // 64*576 bf16  = 73728 B
    float*          sums  = (float*)(ws + 5382144);             // 64 f32
    float*          sumsq = (float*)(ws + 5382144 + 256);       // 64 f32
    float*          y     = (float*)d_out;                      // (8,64,96,96) f32

    offset_conv_kernel<<<NPIX / 256, 256, 0, stream>>>(x, w_off, b_off, off);
    wdef_to_bf16_kernel<<<(COUT * KDIM + 255) / 256, 256, 0, stream>>>(w_def, wbf);
    zero_kernel<<<1, 128, 0, stream>>>(sums, 128);              // sums + sumsq contiguous
    deform_gemm_kernel<<<NPIX / 32, 128, 0, stream>>>(x, off, wbf, b_def, y);
    bn_stats_kernel<<<BB * COUT, 256, 0, stream>>>(y, sums, sumsq);
    bn_relu_kernel<<<(BB * COUT * HW) / 256, 256, 0, stream>>>(y, sums, sumsq, gamma, beta);
}
